// FluxGNN_15917148799344
// MI455X (gfx1250) — matrix operand order repaired and checked by reading, not compiled, not run on hardware
//
#include <hip/hip_runtime.h>
#include <math.h>

#define NN_ 50000
#define NE_ 800000
#define GAMMA_ 1.4f
#define DT_MAX_ 0.015f

typedef __bf16 bf16;
typedef __attribute__((ext_vector_type(16))) __bf16 v16bf;
typedef __attribute__((ext_vector_type(8)))  __bf16 v8bf;
typedef __attribute__((ext_vector_type(8)))  float  v8f;
typedef int v4i __attribute__((__vector_size__(16)));
typedef unsigned int u32x4 __attribute__((ext_vector_type(4)));
typedef int i32x4 __attribute__((ext_vector_type(4)));
typedef int i32x8 __attribute__((ext_vector_type(8)));
#define GAS_ __attribute__((address_space(1)))
#define LAS_ __attribute__((address_space(3)))

#if defined(__gfx1250__) && __has_builtin(__builtin_amdgcn_global_load_async_to_lds_b128) && __has_builtin(__builtin_amdgcn_s_wait_asynccnt)
#define USE_ASYNC_G2L 1
#else
#define USE_ASYNC_G2L 0
#endif
#if defined(__gfx1250__) && __has_builtin(__builtin_amdgcn_tensor_load_to_lds) && __has_builtin(__builtin_amdgcn_s_wait_tensorcnt)
#define USE_TDM 1
#else
#define USE_TDM 0
#endif

struct Stats {
  const float *rho_mean, *rho_std, *e_mean, *e_std, *p_mean, *p_std, *rhou_mean, *rhou_std;
};
struct NodeW {
  const bf16 *pn1, *pn2, *p1a, *p1b, *p2a, *p2b;
  const float *bn1, *bn2, *b1a, *b1b, *b2a, *b2b;
};
struct EdgeW {
  const bf16 *pm1, *pm2;
  const float *bm1, *bm2;
  const bf16 *ph0, *ph1, *ph2;
  const float *bh0, *bh1, *bh2;
  const bf16 *po0, *po1, *po2;
  const float *bo0, *bo1, *bo2;
  const float *s;
};

// branch-free GELU (tanh form), uses hardware v_exp_f32
__device__ __forceinline__ float geluf(float x) {
  float u = x * (0.7978845608f + 0.0356774081f * x * x);
  float e = __expf(2.0f * u);
  float t = 1.0f - 2.0f / (e + 1.0f);
  return 0.5f * x * (1.0f + t);
}
__device__ __forceinline__ v8f zero8() {
  v8f v;
#pragma unroll
  for (int i = 0; i < 8; i++) v[i] = 0.0f;
  return v;
}
__device__ __forceinline__ v8f wmma_bf(v16bf a, v16bf b, v8f c) {
  return __builtin_amdgcn_wmma_f32_16x16x32_bf16(false, a, false, b, (short)0, c, false, false);
}
// packed B fragments: ((kblk*nn + nblk)*32 + lane) * 16 bf16 elements (global or LDS)
__device__ __forceinline__ v16bf loadB(const bf16* packed, int nn, int kblk, int nblk, int lane) {
  return *(const v16bf*)(packed + ((size_t)((kblk * nn + nblk) * 32 + lane) << 4));
}
// A fragment from bf16 LDS tile (row-major, stride elements)
__device__ __forceinline__ v16bf loadA_lds(const bf16* t, int stride, int colofs, int kblk, int lane) {
  int m = lane & 15;
  int kh = (lane & 16) ? 8 : 0;
  const bf16* p = t + m * stride + colofs + kblk * 32 + kh;
  v8bf lo = *(const v8bf*)(p);
  v8bf hi = *(const v8bf*)(p + 16);
  v16bf a;
#pragma unroll
  for (int i = 0; i < 8; i++) { a[i] = lo[i]; a[8 + i] = hi[i]; }
  return a;
}
__device__ __forceinline__ void stageD_lds(bf16* t, int stride, int colofs, int nblk, v8f acc,
                                           const float* bias, int nbias, bool act, int lane) {
  int c = lane & 15;
  int mb = (lane & 16) ? 8 : 0;
  int col = nblk * 16 + c;
  float b = (col < nbias) ? bias[col] : 0.0f;
#pragma unroll
  for (int j = 0; j < 8; j++) {
    float v = acc[j] + b;
    if (act) v = geluf(v);
    t[(mb + j) * stride + colofs + col] = (bf16)v;
  }
}
__device__ __forceinline__ void stageOut_lds(float* t, int stride, int colofs, v8f acc,
                                             const float* bias, int ncols, int lane) {
  int c = lane & 15;
  int mb = (lane & 16) ? 8 : 0;
  if (c < ncols) {
#pragma unroll
    for (int j = 0; j < 8; j++) t[(mb + j) * stride + colofs + c] = acc[j] + bias[c];
  }
}
// bulk weight staging global -> LDS (per-lane async-to-LDS fallback path)
__device__ __forceinline__ void stageW(const bf16* __restrict__ gsrc, bf16* ldst,
                                       int bytes, int tid, int nth) {
#if USE_ASYNC_G2L
  for (int ofs = tid * 16; ofs < bytes; ofs += nth * 16) {
    __builtin_amdgcn_global_load_async_to_lds_b128(
        (GAS_ v4i*)((char*)const_cast<bf16*>(gsrc) + ofs),
        (LAS_ v4i*)((char*)ldst + ofs), 0, 0);
  }
#else
  for (int ofs = tid * 16; ofs < bytes; ofs += nth * 16) {
    *(uint4*)((char*)ldst + ofs) = *(const uint4*)((const char*)gsrc + ofs);
  }
#endif
}
__device__ __forceinline__ void asyncWaitAll() {
#if USE_ASYNC_G2L
  __builtin_amdgcn_s_wait_asynccnt(0);
#endif
}

#if USE_TDM
// TDM: DMA a contiguous weight blob (bytes multiple of 2048) as a 2D tile of
// 8-byte elements: 256 elements per row, bytes/2048 rows. D# per ISA 8.3/8.4.
__device__ __forceinline__ void tdm_load(const void* gsrc, void* lds, int bytes) {
  unsigned lofs = (unsigned)(size_t)(LAS_ char*)lds;       // LDS byte address
  unsigned long long ga = (unsigned long long)(size_t)gsrc;
  int rows = bytes >> 11;
  u32x4 g0;
  g0[0] = 1u;                                              // count=1, user D#
  g0[1] = lofs;                                            // lds_addr
  g0[2] = (unsigned)(ga & 0xFFFFFFFFu);                    // global_addr[31:0]
  g0[3] = (unsigned)((ga >> 32) & 0x1FFFFFFu) | (2u << 30);// global_addr[56:32] | type=2
  i32x8 g1;
  g1[0] = (3 << 16);              // wg_mask=0, data_size=8B, no pad/iterate
  g1[1] = (256 << 16);            // tensor_dim0[15:0]=256 (bits 63:48)
  g1[2] = (rows << 16);           // tensor_dim0[31:16]=0, tensor_dim1[15:0]=rows
  g1[3] = (256 << 16);            // tensor_dim1[31:16]=0, tile_dim0=256
  g1[4] = rows;                   // tile_dim1=rows, tile_dim2=0
  g1[5] = 256;                    // tensor_dim0_stride[31:0]=256
  g1[6] = 0;
  g1[7] = 0;                      // tensor_dim1_stride=0 (unused for 2D tile)
  i32x4 z4 = {0, 0, 0, 0};
#if __clang_major__ >= 23
  i32x8 z8 = {0, 0, 0, 0, 0, 0, 0, 0};
  __builtin_amdgcn_tensor_load_to_lds(g0, g1, z4, z4, z8, 0);
#else
  __builtin_amdgcn_tensor_load_to_lds(g0, g1, z4, z4, 0);
#endif
}
#endif
#define WAVE_SYNC() __builtin_amdgcn_wave_barrier()

// ---------------- weight repack: fp32 (K x Ncols row-major) -> bf16 B fragments ----------
__global__ void repack_w_kernel(const float* __restrict__ W, bf16* __restrict__ out,
                                int K, int Ncols, int Kpad, int Npad) {
  int t = blockIdx.x * blockDim.x + threadIdx.x;
  int nn = Npad >> 4;
  int total = (Kpad >> 5) * nn * 32;
  if (t >= total) return;
  int lane = t & 31;
  int fid = t >> 5;
  int nblk = fid % nn;
  int kblk = fid / nn;
  int c = lane & 15;
  int kb2 = (lane & 16) ? 8 : 0;    // B layout: lanes 0-15 -> K 0..7,16..23, lanes 16-31 -> K 8..15,24..31
  int col = nblk * 16 + c;
  v16bf frag;
#pragma unroll
  for (int i = 0; i < 16; i++) {
    int k = kblk * 32 + kb2 + (i < 8 ? i : i + 8);
    frag[i] = (bf16)((k < K && col < Ncols) ? W[(size_t)k * Ncols + col] : 0.0f);
  }
  *((v16bf*)out + t) = frag;
}

// ---------------- mean(r) reduction ----------------
__global__ void reduce_r_kernel(const float* __restrict__ ea, float* __restrict__ sumr) {
  int i = blockIdx.x * blockDim.x + threadIdx.x;
  float v = (i < NE_) ? ea[(size_t)i * 3 + 2] : 0.0f;
#pragma unroll
  for (int o = 16; o > 0; o >>= 1) v += __shfl_xor(v, o, 32);
  if ((threadIdx.x & 31) == 0) atomicAdd(sumr, v);
}

// ---------------- encoder: x(16) -> 128 gelu -> 4, then eos/p_n ----------------
__global__ void encoder_kernel(const float* __restrict__ x,
                               const bf16* __restrict__ pW0, const float* __restrict__ b0,
                               const bf16* __restrict__ pW1, const float* __restrict__ b1,
                               float* __restrict__ node_u, Stats st) {
  __shared__ bf16 tile[16 * 128];
  __shared__ float outU[16 * 4];
  int lane = threadIdx.x;
  int t0 = blockIdx.x * 16;
  int m = lane & 15;
  int kh = (lane & 16) ? 8 : 0;
  v16bf A;
  {
    const float* p = x + (size_t)(t0 + m) * 16 + kh;
#pragma unroll
    for (int i = 0; i < 8; i++) { A[i] = (bf16)p[i]; A[8 + i] = (bf16)0.0f; }
  }
#pragma unroll
  for (int nblk = 0; nblk < 8; nblk++) {
    v8f acc = zero8();
    acc = wmma_bf(A, loadB(pW0, 8, 0, nblk, lane), acc);
    stageD_lds(tile, 128, 0, nblk, acc, b0, 128, true, lane);
  }
  __syncthreads();
  v16bf F[4];
#pragma unroll
  for (int kb = 0; kb < 4; kb++) F[kb] = loadA_lds(tile, 128, 0, kb, lane);
  v8f acc = zero8();
#pragma unroll
  for (int kb = 0; kb < 4; kb++) acc = wmma_bf(F[kb], loadB(pW1, 1, kb, 0, lane), acc);
  stageOut_lds(outU, 4, 0, acc, b1, 4, lane);
  __syncthreads();
  if (lane < 16) {
    int n = t0 + lane;
    float rho = outU[lane * 4 + 0], e = outU[lane * 4 + 1];
    float m0 = outU[lane * 4 + 2], m1 = outU[lane * 4 + 3];
    float rd = rho * st.rho_std[0] + st.rho_mean[0];
    float ed = e * st.e_std[0] + st.e_mean[0];
    float md0 = m0 * st.rhou_std[0] + st.rhou_mean[0];
    float md1 = m1 * st.rhou_std[0] + st.rhou_mean[0];
    float ke = 0.5f * (md0 * md0 + md1 * md1) / fmaxf(rd, 1e-6f);
    float p = (GAMMA_ - 1.0f) * (ed - ke);
    float pn = (p - st.p_mean[0]) / st.p_std[0];
    float* o = node_u + (size_t)n * 8;
    o[0] = rho; o[1] = e; o[2] = pn; o[3] = m0; o[4] = m1; o[5] = 0.f; o[6] = 0.f; o[7] = 0.f;
  }
}

// ---------------- per-node stage (persistent, weights in LDS via TDM) ----------------
#define NWAVES 4
__global__ void __launch_bounds__(32 * NWAVES) node_stage_kernel(
    const float* __restrict__ node_u, NodeW w, bf16* __restrict__ g, int numTiles) {
  __shared__ bf16 wN1[32 * 64], wN2[64 * 128];
  __shared__ bf16 w1A[128 * 64], w1B[64 * 128], w2A[128 * 64], w2B[64 * 128];
  __shared__ bf16 tileHS[NWAVES * 16 * 128];
  __shared__ bf16 tileBS[NWAVES * 16 * 64];
  int tid = threadIdx.x;
#if USE_TDM
  if (tid < 32) {  // wave 0 issues all TDM descriptor loads
    tdm_load(w.pn1, wN1, 32 * 64 * 2);
    tdm_load(w.pn2, wN2, 64 * 128 * 2);
    tdm_load(w.p1a, w1A, 128 * 64 * 2);
    tdm_load(w.p1b, w1B, 64 * 128 * 2);
    tdm_load(w.p2a, w2A, 128 * 64 * 2);
    tdm_load(w.p2b, w2B, 64 * 128 * 2);
    __builtin_amdgcn_s_wait_tensorcnt(0);
  }
#else
  stageW(w.pn1, wN1, 32 * 64 * 2, tid, 32 * NWAVES);
  stageW(w.pn2, wN2, 64 * 128 * 2, tid, 32 * NWAVES);
  stageW(w.p1a, w1A, 128 * 64 * 2, tid, 32 * NWAVES);
  stageW(w.p1b, w1B, 64 * 128 * 2, tid, 32 * NWAVES);
  stageW(w.p2a, w2A, 128 * 64 * 2, tid, 32 * NWAVES);
  stageW(w.p2b, w2B, 64 * 128 * 2, tid, 32 * NWAVES);
  asyncWaitAll();
#endif
  __syncthreads();
  int wave = tid >> 5, lane = tid & 31;
  bf16* tileH = tileHS + wave * (16 * 128);
  bf16* tileB = tileBS + wave * (16 * 64);
  int m = lane & 15;
  int kh = (lane & 16) ? 8 : 0;
  for (int t = blockIdx.x * NWAVES + wave; t < numTiles; t += gridDim.x * NWAVES) {
    int t0 = t * 16;
    v16bf A;
    {
      const float* p = node_u + (size_t)(t0 + m) * 8;
#pragma unroll
      for (int i = 0; i < 8; i++) {
        int k = kh + i;
        A[i] = (bf16)((k < 5) ? p[k] : 0.0f);
        A[8 + i] = (bf16)0.0f;
      }
    }
    // phi_node layer1: 5(->32) x 64, gelu
#pragma unroll
    for (int nblk = 0; nblk < 4; nblk++) {
      v8f acc = zero8();
      acc = wmma_bf(A, loadB(wN1, 4, 0, nblk, lane), acc);
      stageD_lds(tileB, 64, 0, nblk, acc, w.bn1, 64, true, lane);
    }
    WAVE_SYNC();
    v16bf P0 = loadA_lds(tileB, 64, 0, 0, lane);
    v16bf P1 = loadA_lds(tileB, 64, 0, 1, lane);
    WAVE_SYNC();
    // phi_node layer2: 64 x 128 -> h
#pragma unroll
    for (int nblk = 0; nblk < 8; nblk++) {
      v8f acc = zero8();
      acc = wmma_bf(P0, loadB(wN2, 8, 0, nblk, lane), acc);
      acc = wmma_bf(P1, loadB(wN2, 8, 1, nblk, lane), acc);
      stageD_lds(tileH, 128, 0, nblk, acc, w.bn2, 128, false, lane);
    }
    WAVE_SYNC();
    v16bf H[4];
#pragma unroll
    for (int kb = 0; kb < 4; kb++) H[kb] = loadA_lds(tileH, 128, 0, kb, lane);
    // phi1 hidden: 128 x 64, gelu
#pragma unroll
    for (int nblk = 0; nblk < 4; nblk++) {
      v8f acc = zero8();
#pragma unroll
      for (int kb = 0; kb < 4; kb++) acc = wmma_bf(H[kb], loadB(w1A, 4, kb, nblk, lane), acc);
      stageD_lds(tileB, 64, 0, nblk, acc, w.b1a, 64, true, lane);
    }
    WAVE_SYNC();
    v16bf G1a = loadA_lds(tileB, 64, 0, 0, lane);
    v16bf G1b = loadA_lds(tileB, 64, 0, 1, lane);
    WAVE_SYNC();
    // phi2 hidden: 128 x 64, gelu
#pragma unroll
    for (int nblk = 0; nblk < 4; nblk++) {
      v8f acc = zero8();
#pragma unroll
      for (int kb = 0; kb < 4; kb++) acc = wmma_bf(H[kb], loadB(w2A, 4, kb, nblk, lane), acc);
      stageD_lds(tileB, 64, 0, nblk, acc, w.b2a, 64, true, lane);
    }
    WAVE_SYNC();
    v16bf G2a = loadA_lds(tileB, 64, 0, 0, lane);
    v16bf G2b = loadA_lds(tileB, 64, 0, 1, lane);
    // g = phi1_2 + phi2_2 (shared accumulator chain), store bf16
#pragma unroll
    for (int nblk = 0; nblk < 8; nblk++) {
      v8f acc = zero8();
      acc = wmma_bf(G1a, loadB(w1B, 8, 0, nblk, lane), acc);
      acc = wmma_bf(G1b, loadB(w1B, 8, 1, nblk, lane), acc);
      acc = wmma_bf(G2a, loadB(w2B, 8, 0, nblk, lane), acc);
      acc = wmma_bf(G2b, loadB(w2B, 8, 1, nblk, lane), acc);
      int c = lane & 15;
      int mb = (lane & 16) ? 8 : 0;
      int col = nblk * 16 + c;
      float bb = w.b1b[col] + w.b2b[col];
#pragma unroll
      for (int j = 0; j < 8; j++)
        g[(size_t)(t0 + mb + j) * 128 + col] = (bf16)(acc[j] + bb);
    }
    WAVE_SYNC();
  }
}

// ---------------- eps = phi_edge(r): 1(->32) -> 32 gelu -> 32 ----------------
__global__ void eps_kernel(const float* __restrict__ ea,
                           const bf16* __restrict__ pW1, const float* __restrict__ b1,
                           const bf16* __restrict__ pW2, const float* __restrict__ b2,
                           bf16* __restrict__ eps) {
  __shared__ bf16 tile[16 * 32];
  int lane = threadIdx.x;
  int t0 = blockIdx.x * 16;
  int m = lane & 15;
  v16bf A;
  {
#pragma unroll
    for (int i = 0; i < 16; i++) A[i] = (bf16)0.0f;
    if ((lane & 16) == 0) A[0] = (bf16)ea[(size_t)(t0 + m) * 3 + 2];
  }
#pragma unroll
  for (int nblk = 0; nblk < 2; nblk++) {
    v8f acc = zero8();
    acc = wmma_bf(A, loadB(pW1, 2, 0, nblk, lane), acc);
    stageD_lds(tile, 32, 0, nblk, acc, b1, 32, true, lane);
  }
  __syncthreads();
  v16bf F = loadA_lds(tile, 32, 0, 0, lane);
#pragma unroll
  for (int nblk = 0; nblk < 2; nblk++) {
    v8f acc = zero8();
    acc = wmma_bf(F, loadB(pW2, 2, 0, nblk, lane), acc);
    int c = lane & 15;
    int mb = (lane & 16) ? 8 : 0;
    int col = nblk * 16 + c;
    float bb = b2[col];
#pragma unroll
    for (int j = 0; j < 8; j++) eps[(size_t)(t0 + mb + j) * 32 + col] = (bf16)(acc[j] + bb);
  }
}

// ---------------- edge stage (persistent, all layer weights in LDS via TDM) ----------------
#define EWAVES 4
__global__ void __launch_bounds__(32 * EWAVES) edge_stage_kernel(
    const bf16* __restrict__ g, const bf16* __restrict__ eps, const float* __restrict__ ea,
    const int* __restrict__ ei_src, const int* __restrict__ ei_dst,
    const float* __restrict__ sumr, EdgeW w, float* __restrict__ delta, int numTiles) {
  __shared__ bf16 wPM1[160 * 128];
  __shared__ bf16 wPM2[128 * 128];
  __shared__ bf16 wPH[3 * 128 * 64];
  __shared__ bf16 wPO[3 * 64 * 16];
  __shared__ bf16 tileS[EWAVES * 16 * 192];
  __shared__ float out4S[EWAVES * 64];
  __shared__ int idxS[EWAVES * 32];
  int tid = threadIdx.x;
#if USE_TDM
  if (tid < 32) {  // wave 0 issues all TDM descriptor loads
    tdm_load(w.pm1, wPM1, 160 * 128 * 2);
    tdm_load(w.pm2, wPM2, 128 * 128 * 2);
    tdm_load(w.ph0, wPH, 128 * 64 * 2);
    tdm_load(w.ph1, wPH + 128 * 64, 128 * 64 * 2);
    tdm_load(w.ph2, wPH + 2 * 128 * 64, 128 * 64 * 2);
    tdm_load(w.po0, wPO, 64 * 16 * 2);
    tdm_load(w.po1, wPO + 64 * 16, 64 * 16 * 2);
    tdm_load(w.po2, wPO + 2 * 64 * 16, 64 * 16 * 2);
    __builtin_amdgcn_s_wait_tensorcnt(0);
  }
#else
  {
    const int NTH = 32 * EWAVES;
    stageW(w.pm1, wPM1, 160 * 128 * 2, tid, NTH);
    stageW(w.pm2, wPM2, 128 * 128 * 2, tid, NTH);
    stageW(w.ph0, wPH, 128 * 64 * 2, tid, NTH);
    stageW(w.ph1, wPH + 128 * 64, 128 * 64 * 2, tid, NTH);
    stageW(w.ph2, wPH + 2 * 128 * 64, 128 * 64 * 2, tid, NTH);
    stageW(w.po0, wPO, 64 * 16 * 2, tid, NTH);
    stageW(w.po1, wPO + 64 * 16, 64 * 16 * 2, tid, NTH);
    stageW(w.po2, wPO + 2 * 64 * 16, 64 * 16 * 2, tid, NTH);
    asyncWaitAll();
  }
#endif
  __syncthreads();
  int wave = tid >> 5, lane = tid & 31;
  bf16* tile = tileS + wave * (16 * 192);
  float* out4 = out4S + wave * 64;
  int* sIdx = idxS + wave * 32;
  int* dIdx = sIdx + 16;
  int m = lane & 15;
  int kh = (lane & 16) ? 8 : 0;
  // uniform flux scale
  float mean_r = sumr[0] / (float)NE_;
  float area = mean_r * mean_r;
  float dt = DT_MAX_ / (1.0f + __expf(-w.s[0]));
  float scale = dt / area;
  for (int t = blockIdx.x * EWAVES + wave; t < numTiles; t += gridDim.x * EWAVES) {
    int t0 = t * 16;
    if (lane < 16) {
      sIdx[lane] = ei_src[t0 + lane];
      dIdx[lane] = ei_dst[t0 + lane];
    }
    __builtin_prefetch(ea + (size_t)(t0 + 16) * 3, 0, 3);
    WAVE_SYNC();
    int e = t0 + m;
    v16bf A[5];
    {
      const bf16* ps = g + (size_t)sIdx[m] * 128;
      const bf16* pd = g + (size_t)dIdx[m] * 128;
#pragma unroll
      for (int kb = 0; kb < 4; kb++) {
        v8bf lo1 = *(const v8bf*)(ps + kb * 32 + kh);
        v8bf hi1 = *(const v8bf*)(ps + kb * 32 + kh + 16);
        v8bf lo2 = *(const v8bf*)(pd + kb * 32 + kh);
        v8bf hi2 = *(const v8bf*)(pd + kb * 32 + kh + 16);
#pragma unroll
        for (int i = 0; i < 8; i++) {
          A[kb][i]     = (bf16)((float)lo1[i] + (float)lo2[i]);
          A[kb][8 + i] = (bf16)((float)hi1[i] + (float)hi2[i]);
        }
      }
      const bf16* pe = eps + (size_t)e * 32;
      v8bf lo = *(const v8bf*)(pe + kh);
      v8bf hi = *(const v8bf*)(pe + kh + 16);
#pragma unroll
      for (int i = 0; i < 8; i++) { A[4][i] = lo[i]; A[4][8 + i] = hi[i]; }
    }
    // phi_msg L1: 160 x 128, gelu
#pragma unroll
    for (int nblk = 0; nblk < 8; nblk++) {
      v8f acc = zero8();
#pragma unroll
      for (int kb = 0; kb < 5; kb++) acc = wmma_bf(A[kb], loadB(wPM1, 8, kb, nblk, lane), acc);
      stageD_lds(tile, 128, 0, nblk, acc, w.bm1, 128, true, lane);
    }
    WAVE_SYNC();
    v16bf M1[4];
#pragma unroll
    for (int kb = 0; kb < 4; kb++) M1[kb] = loadA_lds(tile, 128, 0, kb, lane);
    WAVE_SYNC();
    // phi_msg L2: 128 x 128 -> m
#pragma unroll
    for (int nblk = 0; nblk < 8; nblk++) {
      v8f acc = zero8();
#pragma unroll
      for (int kb = 0; kb < 4; kb++) acc = wmma_bf(M1[kb], loadB(wPM2, 8, kb, nblk, lane), acc);
      stageD_lds(tile, 128, 0, nblk, acc, w.bm2, 128, false, lane);
    }
    WAVE_SYNC();
    v16bf M2[4];
#pragma unroll
    for (int kb = 0; kb < 4; kb++) M2[kb] = loadA_lds(tile, 128, 0, kb, lane);
    WAVE_SYNC();
    // psi hidden layers: 3 heads of 128 -> 64, gelu, packed side-by-side (stride 192)
    const float* BH[3] = {w.bh0, w.bh1, w.bh2};
#pragma unroll
    for (int h = 0; h < 3; h++) {
#pragma unroll
      for (int nblk = 0; nblk < 4; nblk++) {
        v8f acc = zero8();
#pragma unroll
        for (int kb = 0; kb < 4; kb++)
          acc = wmma_bf(M2[kb], loadB(wPH + h * 128 * 64, 4, kb, nblk, lane), acc);
        stageD_lds(tile, 192, h * 64, nblk, acc, BH[h], 64, true, lane);
      }
    }
    WAVE_SYNC();
    // psi outputs: 64 -> {1,1,2}
    const float* BO[3] = {w.bo0, w.bo1, w.bo2};
    const int NCOLS[3] = {1, 1, 2};
    const int COFS[3] = {0, 1, 2};
#pragma unroll
    for (int h = 0; h < 3; h++) {
      v16bf Ha = loadA_lds(tile, 192, h * 64, 0, lane);
      v16bf Hb = loadA_lds(tile, 192, h * 64, 1, lane);
      v8f acc = zero8();
      acc = wmma_bf(Ha, loadB(wPO + h * 64 * 16, 1, 0, 0, lane), acc);
      acc = wmma_bf(Hb, loadB(wPO + h * 64 * 16, 1, 1, 0, lane), acc);
      stageOut_lds(out4, 4, COFS[h], acc, BO[h], NCOLS[h], lane);
    }
    WAVE_SYNC();
    if (lane < 16) {
      float arho = out4[lane * 4 + 0];
      float ae = out4[lane * 4 + 1];
      float am0 = out4[lane * 4 + 2];
      float am1 = out4[lane * 4 + 3];
      int ee = t0 + lane;
      float dx = ea[(size_t)ee * 3 + 0];
      float dy = ea[(size_t)ee * 3 + 1];
      float r = ea[(size_t)ee * 3 + 2];
      float inv = 1.0f / (r + 1e-12f);
      float nx = dx * inv, ny = dy * inv;
      float tx = -ny, ty = nx;
      float nn = nx * nx + ny * ny;
      float fr = arho * nn * r;
      float fe = ae * nn * r;
      float fm0 = (am0 * nx + am1 * tx) * r;
      float fm1 = (am0 * ny + am1 * ty) * r;
      float c0 = scale * fr, c1 = scale * fe, c2 = scale * fm0, c3 = scale * fm1;
      int sI = sIdx[lane], dI = dIdx[lane];
      atomicAdd(delta + (size_t)sI * 4 + 0, -c0);
      atomicAdd(delta + (size_t)sI * 4 + 1, -c1);
      atomicAdd(delta + (size_t)sI * 4 + 2, -c2);
      atomicAdd(delta + (size_t)sI * 4 + 3, -c3);
      atomicAdd(delta + (size_t)dI * 4 + 0, c0);
      atomicAdd(delta + (size_t)dI * 4 + 1, c1);
      atomicAdd(delta + (size_t)dI * 4 + 2, c2);
      atomicAdd(delta + (size_t)dI * 4 + 3, c3);
    }
    WAVE_SYNC();
  }
}

// ---------------- node update: u += delta, recompute p_n ----------------
__global__ void node_update_kernel(float* __restrict__ node_u, const float* __restrict__ delta,
                                   Stats st) {
  int n = blockIdx.x * blockDim.x + threadIdx.x;
  if (n >= NN_) return;
  float* u = node_u + (size_t)n * 8;
  const float* d = delta + (size_t)n * 4;
  float rho = u[0] + d[0];
  float e = u[1] + d[1];
  float m0 = u[3] + d[2];
  float m1 = u[4] + d[3];
  float rd = rho * st.rho_std[0] + st.rho_mean[0];
  float ed = e * st.e_std[0] + st.e_mean[0];
  float md0 = m0 * st.rhou_std[0] + st.rhou_mean[0];
  float md1 = m1 * st.rhou_std[0] + st.rhou_mean[0];
  float ke = 0.5f * (md0 * md0 + md1 * md1) / fmaxf(rd, 1e-6f);
  float p = (GAMMA_ - 1.0f) * (ed - ke);
  float pn = (p - st.p_mean[0]) / st.p_std[0];
  u[0] = rho; u[1] = e; u[2] = pn; u[3] = m0; u[4] = m1;
}

__global__ void writeout_kernel(const float* __restrict__ node_u, float* __restrict__ out) {
  int i = blockIdx.x * blockDim.x + threadIdx.x;
  if (i >= NN_ * 5) return;
  int n = i / 5, k = i % 5;
  out[i] = node_u[(size_t)n * 8 + k];
}

extern "C" void kernel_launch(void* const* d_in, const int* in_sizes, int n_in,
                              void* d_out, int out_size, void* d_ws, size_t ws_size,
                              hipStream_t stream) {
  (void)in_sizes; (void)out_size;
  if (n_in < 179) return;  // flatten: x, edge_attr, 8 stats, encoder(4), 4*41 layer leaves, edge_index
  const float* x = (const float*)d_in[0];
  const float* ea = (const float*)d_in[1];
  Stats st;
  st.rho_mean = (const float*)d_in[2]; st.rho_std = (const float*)d_in[3];
  st.e_mean = (const float*)d_in[4];   st.e_std = (const float*)d_in[5];
  st.p_mean = (const float*)d_in[6];   st.p_std = (const float*)d_in[7];
  st.rhou_mean = (const float*)d_in[8]; st.rhou_std = (const float*)d_in[9];
  const int* ei = (const int*)d_in[178];

  // ---- workspace layout ----
  char* base = (char*)d_ws;
  size_t off = 0;
  auto alloc = [&](size_t bytes) -> void* {
    void* p = base + off;
    off = (off + bytes + 255) & ~(size_t)255;
    return p;
  };
  float* node_u = (float*)alloc((size_t)NN_ * 8 * 4);
  bf16* g = (bf16*)alloc((size_t)NN_ * 128 * 2);
  bf16* eps = (bf16*)alloc((size_t)NE_ * 32 * 2);
  float* delta = (float*)alloc((size_t)NN_ * 4 * 4);
  float* sumr = (float*)alloc(256);
  bf16* pkE0 = (bf16*)alloc((size_t)32 * 128 * 2);
  bf16* pkE1 = (bf16*)alloc((size_t)128 * 16 * 2);
  bf16 *pn1[4], *pn2[4], *p1a[4], *p1b[4], *p2a[4], *p2b[4];
  bf16 *pe1[4], *pe2[4], *pm1[4], *pm2[4];
  bf16 *phr[4], *por[4], *phe[4], *poe[4], *phu[4], *pou[4];
  for (int l = 0; l < 4; l++) {
    pn1[l] = (bf16*)alloc((size_t)32 * 64 * 2);
    pn2[l] = (bf16*)alloc((size_t)64 * 128 * 2);
    p1a[l] = (bf16*)alloc((size_t)128 * 64 * 2);
    p1b[l] = (bf16*)alloc((size_t)64 * 128 * 2);
    p2a[l] = (bf16*)alloc((size_t)128 * 64 * 2);
    p2b[l] = (bf16*)alloc((size_t)64 * 128 * 2);
    pe1[l] = (bf16*)alloc((size_t)32 * 32 * 2);
    pe2[l] = (bf16*)alloc((size_t)32 * 32 * 2);
    pm1[l] = (bf16*)alloc((size_t)160 * 128 * 2);
    pm2[l] = (bf16*)alloc((size_t)128 * 128 * 2);
    phr[l] = (bf16*)alloc((size_t)128 * 64 * 2);
    por[l] = (bf16*)alloc((size_t)64 * 16 * 2);
    phe[l] = (bf16*)alloc((size_t)128 * 64 * 2);
    poe[l] = (bf16*)alloc((size_t)64 * 16 * 2);
    phu[l] = (bf16*)alloc((size_t)128 * 64 * 2);
    pou[l] = (bf16*)alloc((size_t)64 * 16 * 2);
  }
  if (off > ws_size) return;

  auto repack = [&](const void* W, void* out, int K, int Nc, int Kp, int Np) {
    int total = (Kp / 32) * (Np / 16) * 32;
    repack_w_kernel<<<(total + 255) / 256, 256, 0, stream>>>((const float*)W, (bf16*)out, K, Nc, Kp, Np);
  };

  (void)hipMemsetAsync(sumr, 0, 4, stream);
  reduce_r_kernel<<<(NE_ + 255) / 256, 256, 0, stream>>>(ea, sumr);

  // ---- repack all weights ----
  repack(d_in[10], pkE0, 16, 128, 32, 128);
  repack(d_in[12], pkE1, 128, 4, 128, 16);
  for (int l = 0; l < 4; l++) {
    int lb = 14 + 41 * l;
    repack(d_in[lb + 0], pn1[l], 5, 64, 32, 64);
    repack(d_in[lb + 2], pn2[l], 64, 128, 64, 128);
    repack(d_in[lb + 4], p1a[l], 128, 64, 128, 64);
    repack(d_in[lb + 6], p1b[l], 64, 128, 64, 128);
    repack(d_in[lb + 8], p2a[l], 128, 64, 128, 64);
    repack(d_in[lb + 10], p2b[l], 64, 128, 64, 128);
    repack(d_in[lb + 12], pe1[l], 1, 32, 32, 32);
    repack(d_in[lb + 14], pe2[l], 32, 32, 32, 32);
    repack(d_in[lb + 16], pm1[l], 160, 128, 160, 128);
    repack(d_in[lb + 18], pm2[l], 128, 128, 128, 128);
    repack(d_in[lb + 20], phr[l], 128, 64, 128, 64);
    repack(d_in[lb + 22], por[l], 64, 1, 64, 16);
    repack(d_in[lb + 24], phe[l], 128, 64, 128, 64);
    repack(d_in[lb + 26], poe[l], 64, 1, 64, 16);
    repack(d_in[lb + 28], phu[l], 128, 64, 128, 64);
    repack(d_in[lb + 30], pou[l], 64, 2, 64, 16);
  }

  // ---- encoder ----
  encoder_kernel<<<NN_ / 16, 32, 0, stream>>>(x, pkE0, (const float*)d_in[11],
                                              pkE1, (const float*)d_in[13], node_u, st);

  // ---- layers ----
  for (int l = 0; l < 4; l++) {
    int lb = 14 + 41 * l;
    NodeW nw;
    nw.pn1 = pn1[l]; nw.pn2 = pn2[l]; nw.p1a = p1a[l]; nw.p1b = p1b[l]; nw.p2a = p2a[l]; nw.p2b = p2b[l];
    nw.bn1 = (const float*)d_in[lb + 1]; nw.bn2 = (const float*)d_in[lb + 3];
    nw.b1a = (const float*)d_in[lb + 5]; nw.b1b = (const float*)d_in[lb + 7];
    nw.b2a = (const float*)d_in[lb + 9]; nw.b2b = (const float*)d_in[lb + 11];
    node_stage_kernel<<<250, 32 * NWAVES, 0, stream>>>(node_u, nw, g, NN_ / 16);

    eps_kernel<<<NE_ / 16, 32, 0, stream>>>(ea, pe1[l], (const float*)d_in[lb + 13],
                                            pe2[l], (const float*)d_in[lb + 15], eps);

    (void)hipMemsetAsync(delta, 0, (size_t)NN_ * 4 * 4, stream);

    EdgeW ew;
    ew.pm1 = pm1[l]; ew.pm2 = pm2[l];
    ew.bm1 = (const float*)d_in[lb + 17]; ew.bm2 = (const float*)d_in[lb + 19];
    ew.ph0 = phr[l]; ew.ph1 = phe[l]; ew.ph2 = phu[l];
    ew.bh0 = (const float*)d_in[lb + 21]; ew.bh1 = (const float*)d_in[lb + 25]; ew.bh2 = (const float*)d_in[lb + 29];
    ew.po0 = por[l]; ew.po1 = poe[l]; ew.po2 = pou[l];
    ew.bo0 = (const float*)d_in[lb + 23]; ew.bo1 = (const float*)d_in[lb + 27]; ew.bo2 = (const float*)d_in[lb + 31];
    ew.s = (const float*)d_in[lb + 32];
    edge_stage_kernel<<<1250, 32 * EWAVES, 0, stream>>>(g, eps, ea, ei, ei + NE_, sumr, ew, delta,
                                                        NE_ / 16);

    node_update_kernel<<<(NN_ + 255) / 256, 256, 0, stream>>>(node_u, delta, st);
  }

  writeout_kernel<<<(NN_ * 5 + 255) / 256, 256, 0, stream>>>(node_u, (float*)d_out);
}
